// CycleConsistencyLoss_28200755265842
// MI455X (gfx1250) — matrix operand order repaired and checked by reading, not compiled, not run on hardware
//
#include <hip/hip_runtime.h>
#include <hip/hip_bf16.h>
#include <stdint.h>

#define DD 128  // embedding dim (fixed by the problem)

typedef __attribute__((ext_vector_type(16))) __bf16 v16bf;
typedef __attribute__((ext_vector_type(8)))  __bf16 v8bf;
typedef __attribute__((ext_vector_type(8)))  float  v8f;

union V16U { v16bf v; v8bf h[2]; };

// Load a 32-byte WMMA operand as two 16-byte pieces (16B alignment guaranteed).
__device__ __forceinline__ v16bf load2x16g(const __bf16* p0, const __bf16* p1) {
  V16U u;
  u.h[0] = *(const v8bf*)p0;
  u.h[1] = *(const v8bf*)p1;
  return u.v;
}

__device__ __forceinline__ v8f wmma_bf16(v16bf a, v16bf b, v8f c) {
  // (neg_a, A, neg_b, B, c_mod, C, reuse_a, reuse_b)
  return __builtin_amdgcn_wmma_f32_16x16x32_bf16(false, a, false, b, (short)0, c,
                                                 false, false);
}

// CDNA5 async global->LDS copy (16B per lane), tracked by ASYNCcnt.
__device__ __forceinline__ void async_load_b128(uint32_t lds_off, const void* gptr) {
  asm volatile("global_load_async_to_lds_b128 %0, %1, off"
               :: "v"(lds_off), "v"((uint64_t)(uintptr_t)gptr)
               : "memory");
}
__device__ __forceinline__ void wait_async0() {
  asm volatile("s_wait_asynccnt 0" ::: "memory");
}
// Generic __shared__ pointer -> 32-bit LDS byte offset (low dword of flat addr).
__device__ __forceinline__ uint32_t lds_off32(const void* p) {
  return (uint32_t)(uintptr_t)p;
}

__device__ __forceinline__ float redmax16(float x) {
#pragma unroll
  for (int o = 1; o < 16; o <<= 1) x = fmaxf(x, __shfl_xor(x, o, 32));
  return x;
}
__device__ __forceinline__ float redsum16(float x) {
#pragma unroll
  for (int o = 1; o < 16; o <<= 1) x += __shfl_xor(x, o, 32);
  return x;
}
__device__ __forceinline__ float redsum32(float x) {
#pragma unroll
  for (int o = 1; o < 32; o <<= 1) x += __shfl_xor(x, o, 32);
  return x;
}

// ---------------------------------------------------------------------------
// Zero the two output scalars (harness poisons d_out).
__global__ void cyc_zero_kernel(float* out) {
  if (threadIdx.x < 2) out[threadIdx.x] = 0.0f;
}

// ---------------------------------------------------------------------------
// Preprocess one row per block: f32 -> bf16 (row-major + transposed) + |x|^2.
__global__ __launch_bounds__(128)
void cyc_prep_kernel(const float* __restrict__ emb, __bf16* __restrict__ rm,
                     __bf16* __restrict__ tr, float* __restrict__ sq, int RL) {
  const int r = blockIdx.x;         // global row index (b*RL + m)
  const int d = threadIdx.x;        // 0..127
  const int b = r / RL;
  const int mloc = r - b * RL;
  float x = emb[(size_t)r * DD + d];
  rm[(size_t)r * DD + d] = static_cast<__bf16>(x);
  tr[((size_t)b * DD + d) * (size_t)RL + mloc] = static_cast<__bf16>(x);
  __shared__ float red[128];
  red[d] = x * x;
  __syncthreads();
#pragma unroll
  for (int s2 = 64; s2 > 0; s2 >>= 1) {
    if (d < s2) red[d] += red[d + s2];
    __syncthreads();
  }
  if (d == 0) sq[r] = red[0];
}

// ---------------------------------------------------------------------------
// Pass 1: flash-style soft nearest neighbor.  Workgroup = 4 waves = 64 src
// rows; the 32-column target tile (both the K^T operand and the V operand)
// is staged once per block into double-buffered LDS with async copies and
// shared by all 4 waves.
//   nn[b,i,:] = softmax_j( (2*q_i.t_j - |t_j|^2)/D , masked ) @ tgt
// Masked src rows are written as 0 (they never affect the final loss).
__global__ __launch_bounds__(128)
void cyc_softnn_kernel(const __bf16* __restrict__ src_bf, const float* __restrict__ src_mask,
                       const __bf16* __restrict__ tgtT, const __bf16* __restrict__ tgt_rm,
                       const float* __restrict__ tgt_sq, const float* __restrict__ tgt_mask,
                       __bf16* __restrict__ nn_out, int Ms, int Nt) {
  const int tid  = threadIdx.x;
  const int lane = tid & 31;
  const int wv   = tid >> 5;
  const int b    = blockIdx.y;
  const int row0 = (blockIdx.x * 4 + wv) * 16;
  const int arow = lane & 15;       // A-layout row index
  const int half = lane >> 4;
  const int koff = half * 8;        // A-layout K offset for this lane-half
  const int col  = arow;            // C-layout column
  const float invD = 1.0f / (float)DD;

  __shared__ __align__(16) __bf16 sB[2][128][32];    // tgtT tile [N=d][K=j]  (2x8KB)
  __shared__ __align__(16) __bf16 sV[2][32][128];    // tgt   tile [j][d]   (2x8KB)
  __shared__ __align__(16) __bf16 lds_p[4][16][32];  // per-wave P round trip

  const __bf16* srcp = src_bf + (size_t)b * Ms * DD;
  const __bf16* tT   = tgtT   + (size_t)b * DD * Nt;   // [D][Nt]
  const __bf16* tR   = tgt_rm + (size_t)b * Nt * DD;   // [Nt][D]

  // Q in WMMA A-layout: 4 K-chunks of 32.
  v16bf qa[4];
  {
    const __bf16* rp = srcp + (size_t)(row0 + arow) * DD;
#pragma unroll
    for (int c = 0; c < 4; ++c)
      qa[c] = load2x16g(rp + 32 * c + koff, rp + 32 * c + 16 + koff);
  }

  v8f o[8];                         // 16x128 f32 accumulator (8 N-chunks)
  float m[8], lsum[8];
#pragma unroll
  for (int t = 0; t < 8; ++t)
#pragma unroll
    for (int e = 0; e < 8; ++e) o[t][e] = 0.0f;
#pragma unroll
  for (int v = 0; v < 8; ++v) { m[v] = -3.0e38f; lsum[v] = 0.0f; }

  // --- async stage of one 32-col tile into buffer nb -----------------------
  auto stage = [&](int nb, int j0n) {
    // V operand: thread t copies row d=t (64B) of tgtT[:, j0n:j0n+32]
    const __bf16* gB = tT + (size_t)tid * Nt + j0n;
    uint32_t lB = lds_off32(&sB[nb][tid][0]);
#pragma unroll
    for (int k = 0; k < 4; ++k) async_load_b128(lB + 16 * k, gB + 8 * k);
    // K^T operand for S: thread t copies 64B of row j0n + (t>>2)
    const int jj = tid >> 2;
    const int ch = (tid & 3) * 32;
    const __bf16* gV = tR + (size_t)(j0n + jj) * DD + ch;
    uint32_t lV = lds_off32(&sV[nb][jj][ch]);
#pragma unroll
    for (int k = 0; k < 4; ++k) async_load_b128(lV + 16 * k, gV + 8 * k);
  };

  stage(0, 0);
  wait_async0();
  __syncthreads();

  int buf = 0;
  for (int j0 = 0; j0 < Nt; j0 += 32) {
    if (j0 + 32 < Nt) stage(buf ^ 1, j0 + 32);   // prefetch next tile

    // ---- S = Q * T^T for two 16-col tiles (K = D = 128, 4 WMMAs each)
    v8f s0, s1;
#pragma unroll
    for (int e = 0; e < 8; ++e) { s0[e] = 0.0f; s1[e] = 0.0f; }
#pragma unroll
    for (int c = 0; c < 4; ++c) {
      const __bf16* bp = &sV[buf][arow][32 * c + koff];   // B: lane = N (j), elems = K
      const __bf16* bq = &sV[buf][16 + arow][32 * c + koff];
      s0 = wmma_bf16(qa[c], load2x16g(bp, bp + 16), s0);
      s1 = wmma_bf16(qa[c], load2x16g(bq, bq + 16), s1);
    }

    // ---- masked score + online softmax update
    float sqt0 = tgt_sq  [(size_t)b * Nt + j0 + col];
    float tm0  = tgt_mask[(size_t)b * Nt + j0 + col];
    float sqt1 = tgt_sq  [(size_t)b * Nt + j0 + 16 + col];
    float tm1  = tgt_mask[(size_t)b * Nt + j0 + 16 + col];
    float p0[8], p1[8], cf[8];
#pragma unroll
    for (int v = 0; v < 8; ++v) {
      float a0 = (tm0 > 0.0f) ? (2.0f * s0[v] - sqt0) * invD : -3.0e38f;
      float a1 = (tm1 > 0.0f) ? (2.0f * s1[v] - sqt1) * invD : -3.0e38f;
      float tmax = redmax16(fmaxf(a0, a1));
      float mn = fmaxf(m[v], tmax);
      cf[v] = __expf(m[v] - mn);
      m[v] = mn;
      p0[v] = __expf(a0 - mn);
      p1[v] = __expf(a1 - mn);
      lsum[v] = lsum[v] * cf[v] + p0[v] + p1[v];
    }
#pragma unroll
    for (int t = 0; t < 8; ++t)
#pragma unroll
      for (int v = 0; v < 8; ++v) o[t][v] *= cf[v];

    // ---- P tile: C-layout -> A-layout via wave-private LDS round trip.
    // DS ops are in-order within a wave; no block barrier needed.
#pragma unroll
    for (int v = 0; v < 8; ++v) {
      lds_p[wv][v + 8 * half][col]      = static_cast<__bf16>(p0[v]);
      lds_p[wv][v + 8 * half][16 + col] = static_cast<__bf16>(p1[v]);
    }
    v16bf pa = load2x16g(&lds_p[wv][arow][koff], &lds_p[wv][arow][16 + koff]);

    // ---- O += P * V (A = 16x32 probs, B = 32 tgt rows x 16 d-cols)
#pragma unroll
    for (int t = 0; t < 8; ++t) {
      const __bf16* vp = &sB[buf][16 * t + arow][koff];  // B: lane = N (d), elems = K
      o[t] = wmma_bf16(pa, load2x16g(vp, vp + 16), o[t]);
    }

    wait_async0();     // own prefetch done
    __syncthreads();   // everyone's prefetch done / everyone done reading buf
    buf ^= 1;
  }

  // finalize: divide by row sums, zero masked rows, store bf16.
#pragma unroll
  for (int v = 0; v < 8; ++v) lsum[v] = redsum16(lsum[v]);
  __bf16* np = nn_out + (size_t)b * Ms * DD;
#pragma unroll
  for (int v = 0; v < 8; ++v) {
    int rg = row0 + v + 8 * half;
    float msk = src_mask[(size_t)b * Ms + rg];
    float scale = (msk > 0.0f) ? (1.0f / lsum[v]) : 0.0f;
#pragma unroll
    for (int t = 0; t < 8; ++t)
      np[(size_t)rg * DD + 16 * t + col] = static_cast<__bf16>(o[t][v] * scale);
  }
}

// ---------------------------------------------------------------------------
// Pass 2: streaming softmax of score(nn, src) keeping only (Sum e, Sum j*e);
// accumulates the cycle loss  sum_i (E[j]-i)^2 * mask / (B*lens[b]).
// Same LDS staging scheme for the (shared) K^T operand, 32 columns per step.
__global__ __launch_bounds__(128)
void cyc_beta_loss_kernel(const __bf16* __restrict__ q_bf, const __bf16* __restrict__ tT_all,
                          const float* __restrict__ t_sq, const float* __restrict__ t_mask,
                          const float* __restrict__ lens, float* __restrict__ out,
                          int Nt /* == Mq */, int B) {
  const int tid  = threadIdx.x;
  const int lane = tid & 31;
  const int b    = blockIdx.y;
  const int row0 = (blockIdx.x * 4 + (tid >> 5)) * 16;
  const int arow = lane & 15;
  const int half = lane >> 4;
  const int koff = half * 8;
  const int col  = arow;
  const float invD = 1.0f / (float)DD;

  __shared__ __align__(16) __bf16 sB[2][128][32];    // srcT tile [K=d][j] (2x8KB)

  const __bf16* qp = q_bf   + (size_t)b * Nt * DD;
  const __bf16* tT = tT_all + (size_t)b * DD * Nt;

  v16bf qa[4];
  {
    const __bf16* rp = qp + (size_t)(row0 + arow) * DD;
#pragma unroll
    for (int c = 0; c < 4; ++c)
      qa[c] = load2x16g(rp + 32 * c + koff, rp + 32 * c + 16 + koff);
  }

  float m[8], lsum[8], jsum[8];
#pragma unroll
  for (int v = 0; v < 8; ++v) { m[v] = -3.0e38f; lsum[v] = 0.0f; jsum[v] = 0.0f; }

  auto stage = [&](int nb, int j0n) {
    const __bf16* gB = tT + (size_t)tid * Nt + j0n;   // thread t = row d
    uint32_t lB = lds_off32(&sB[nb][tid][0]);
#pragma unroll
    for (int k = 0; k < 4; ++k) async_load_b128(lB + 16 * k, gB + 8 * k);
  };

  stage(0, 0);
  wait_async0();
  __syncthreads();

  int buf = 0;
  for (int j0 = 0; j0 < Nt; j0 += 32) {
    if (j0 + 32 < Nt) stage(buf ^ 1, j0 + 32);

    v8f s0, s1;
#pragma unroll
    for (int e = 0; e < 8; ++e) { s0[e] = 0.0f; s1[e] = 0.0f; }
#pragma unroll
    for (int c = 0; c < 4; ++c) {
      v16bf b0, b1;
#pragma unroll
      for (int s = 0; s < 8; ++s) {
        b0[s]     = sB[buf][32 * c + koff + s][arow];
        b0[8 + s] = sB[buf][32 * c + 16 + koff + s][arow];
        b1[s]     = sB[buf][32 * c + koff + s][16 + arow];
        b1[8 + s] = sB[buf][32 * c + 16 + koff + s][16 + arow];
      }
      s0 = wmma_bf16(qa[c], b0, s0);
      s1 = wmma_bf16(qa[c], b1, s1);
    }

    float sqt0 = t_sq  [(size_t)b * Nt + j0 + col];
    float tm0  = t_mask[(size_t)b * Nt + j0 + col];
    float sqt1 = t_sq  [(size_t)b * Nt + j0 + 16 + col];
    float tm1  = t_mask[(size_t)b * Nt + j0 + 16 + col];
    float jv0  = (float)(j0 + col);
    float jv1  = (float)(j0 + 16 + col);
#pragma unroll
    for (int v = 0; v < 8; ++v) {
      float a0 = (tm0 > 0.0f) ? (2.0f * s0[v] - sqt0) * invD : -3.0e38f;
      float a1 = (tm1 > 0.0f) ? (2.0f * s1[v] - sqt1) * invD : -3.0e38f;
      float tmax = redmax16(fmaxf(a0, a1));
      float mn = fmaxf(m[v], tmax);
      float cf = __expf(m[v] - mn);
      m[v] = mn;
      float p0 = __expf(a0 - mn);
      float p1 = __expf(a1 - mn);
      lsum[v] = lsum[v] * cf + p0 + p1;
      jsum[v] = jsum[v] * cf + jv0 * p0 + jv1 * p1;
    }

    wait_async0();
    __syncthreads();
    buf ^= 1;
  }

  float contrib = 0.0f;
  float ls = lens[b];
#pragma unroll
  for (int v = 0; v < 8; ++v) {
    float L = redsum16(lsum[v]);
    float J = redsum16(jsum[v]);
    int rg = row0 + v + 8 * half;
    float msk = t_mask[(size_t)b * Nt + rg];  // query mask == target mask here
    if ((lane & 15) == 0 && msk > 0.0f) {
      float idx = J / L;
      float d = idx - (float)rg;
      contrib += d * d;
    }
  }
  contrib *= 1.0f / ((float)B * ls);
  contrib = redsum32(contrib);
  if (lane == 0) atomicAdd(out, contrib);
}

// ---------------------------------------------------------------------------
extern "C" void kernel_launch(void* const* d_in, const int* in_sizes, int n_in,
                              void* d_out, int out_size, void* d_ws, size_t ws_size,
                              hipStream_t stream) {
  const float* clip_emb  = (const float*)d_in[0];
  const float* clip_mask = (const float*)d_in[1];
  const float* clip_lens = (const float*)d_in[2];
  const float* sent_emb  = (const float*)d_in[3];
  const float* sent_mask = (const float*)d_in[4];
  const float* sent_lens = (const float*)d_in[5];

  const int B  = in_sizes[2];
  const int BM = in_sizes[1];
  const int M  = BM / B;
  const int BN = in_sizes[4];
  const int N  = BN / B;

  char* ws = (char*)d_ws;
  size_t off = 0;
  __bf16* clip_bf = (__bf16*)(ws + off); off += (size_t)BM * DD * 2;
  __bf16* sent_bf = (__bf16*)(ws + off); off += (size_t)BN * DD * 2;
  __bf16* clipT   = (__bf16*)(ws + off); off += (size_t)BM * DD * 2;
  __bf16* sentT   = (__bf16*)(ws + off); off += (size_t)BN * DD * 2;
  float*  sq_clip = (float*)(ws + off);  off += (size_t)BM * 4;
  float*  sq_sent = (float*)(ws + off);  off += (size_t)BN * 4;
  __bf16* nn_clip = (__bf16*)(ws + off); off += (size_t)BM * DD * 2;
  __bf16* nn_sent = (__bf16*)(ws + off); off += (size_t)BN * DD * 2;

  float* out = (float*)d_out;

  cyc_zero_kernel<<<1, 32, 0, stream>>>(out);
  cyc_prep_kernel<<<BM, 128, 0, stream>>>(clip_emb, clip_bf, clipT, sq_clip, M);
  cyc_prep_kernel<<<BN, 128, 0, stream>>>(sent_emb, sent_bf, sentT, sq_sent, N);

  dim3 gc(M / 64, B), gs(N / 64, B);
  // clip -> sent soft-NN, then beta vs clip + loss
  cyc_softnn_kernel<<<gc, 128, 0, stream>>>(clip_bf, clip_mask, sentT, sent_bf,
                                            sq_sent, sent_mask, nn_clip, M, N);
  // sent -> clip soft-NN, then beta vs sent + loss
  cyc_softnn_kernel<<<gs, 128, 0, stream>>>(sent_bf, sent_mask, clipT, clip_bf,
                                            sq_clip, clip_mask, nn_sent, N, M);
  cyc_beta_loss_kernel<<<gc, 128, 0, stream>>>(nn_clip, clipT, sq_clip, clip_mask,
                                               clip_lens, out + 0, M, B);
  cyc_beta_loss_kernel<<<gs, 128, 0, stream>>>(nn_sent, sentT, sq_sent, sent_mask,
                                               sent_lens, out + 1, N, B);
}
